// Resnet101RoIHead_30657476559375
// MI455X (gfx1250) — compile-verified, matrix operand order checked by reading, not run on hardware
//
#include <hip/hip_runtime.h>
#include <hip/hip_bf16.h>

// ---------------------------------------------------------------------------
// R-FCN RoI head for MI455X (gfx1250):
//   1) prep: repack W_cls/W_reg (fp32) -> bf16 WMMA A-fragments,
//      layout [kstep][tile][lane][16] so each kstep is one contiguous 18KB blob
//   2) fused GEMM: [272 x 16384] = W(bf16) x F(bf16 converted on the fly),
//      single pass over the 64MB feature map, v_wmma_f32_16x16x32_bf16,
//      double-buffered LDS + TDM (tensor_load_to_lds) for W fragments
//   3) PSROI pooling: 4000 ROIs x 29 dims x 9 bins x 4 bilinear samples,
//      ds_add_f32 LDS reduction
// Workspace: 272*16384*4 B score maps + 32*18432 B W-fragments ~= 18.5 MB
// ---------------------------------------------------------------------------

typedef __attribute__((ext_vector_type(8)))  float        v8f;
typedef __attribute__((ext_vector_type(16))) __bf16       v16bf;
typedef __attribute__((ext_vector_type(4)))  unsigned int v4u;
typedef __attribute__((ext_vector_type(4)))  int          v4i;
typedef __attribute__((ext_vector_type(8)))  int          v8i;

#define HW       16384   // 128*128 pixels
#define KDIM     1024    // input channels
#define KSTEPS   32      // 1024 / 32
#define TILES    18      // 288 padded output rows (261 used: 189 cls | 72 reg)
#define LDSPITCH 48      // uint16 pitch per pixel row (96 B, 16B-aligned)
#define WSTEP_BYTES (TILES * 32 * 32)   // 18432 B of A-fragments per K-step
#define HAVE_TDM __has_builtin(__builtin_amdgcn_tensor_load_to_lds)

union V16B { uint4 q[2]; v16bf v; };

__device__ __forceinline__ unsigned short f32_bf16(float f) {
  unsigned int u = __float_as_uint(f);
  u += 0x7FFFu + ((u >> 16) & 1u);            // round-to-nearest-even
  return (unsigned short)(u >> 16);
}

// ---------------------------------------------------------------------------
// Kernel 1: repack weights into WMMA 16x32 bf16 A-fragment order.
// wfrag[((ks*18 + tile)*32 + lane)*16 + j]
// lane<16: row=tile*16+lane, K {0..7,16..23}; lane>=16: K {8..15,24..31}
// Combined row space: rows 0..188 = W_cls, rows 192..263 = W_reg, rest = 0.
// ---------------------------------------------------------------------------
__global__ void prep_wfrag(const float* __restrict__ W_cls,
                           const float* __restrict__ W_reg,
                           unsigned short* __restrict__ wfrag) {
  int t = blockIdx.x * blockDim.x + threadIdx.x;
  if (t >= TILES * KSTEPS * 32) return;
  int lane = t & 31;
  int g    = t >> 5;
  int tile = g % TILES;
  int ks   = g / TILES;
  int half = lane >> 4;
  int m    = lane & 15;
  int row  = tile * 16 + m;
  const float* src = nullptr;
  if (row < 189)                      src = W_cls + (size_t)row * KDIM;
  else if (row >= 192 && row < 264)   src = W_reg + (size_t)(row - 192) * KDIM;
  unsigned short* dst = wfrag + (size_t)t * 16;
  int kb = ks * 32;
#pragma unroll
  for (int j = 0; j < 16; ++j) {
    int k = kb + (j >> 3) * 16 + half * 8 + (j & 7);
    float v = src ? src[k] : 0.0f;
    dst[j] = f32_bf16(v);
  }
}

// ---------------------------------------------------------------------------
// Kernel 2: fused GEMM. grid = 256 blocks (64 pixels each), 256 threads.
// Wave w: N sub-tile (w&3)*16 pixels, M group (w>>2) -> 9 tiles of 16 rows.
// Double-buffered LDS for both the feature tile (fp32->bf16 staged) and the
// W-fragment blob (TDM async copy issued one K-step ahead by wave 0).
// ---------------------------------------------------------------------------
__global__ __launch_bounds__(256) void fused_gemm(
    const float* __restrict__ F,               // [1024][16384] fp32
    const unsigned short* __restrict__ wfrag,  // bf16 A-fragments (per-ks blobs)
    const float* __restrict__ b_cls,
    const float* __restrict__ b_reg,
    float* __restrict__ out) {                 // [272][16384] fp32 (ws)
  __shared__ unsigned short ldsF[2][64 * LDSPITCH];       // 2 x 6 KB
  __shared__ unsigned short ldsW[2][WSTEP_BYTES / 2];     // 2 x 18 KB

  const int tid  = threadIdx.x;
  const int wave = tid >> 5;
  const int lane = tid & 31;
  const int hw0  = blockIdx.x * 64;
  const int nsub = wave & 3;
  const int mgrp = wave >> 2;
  const int mt0  = mgrp * 9;                   // tiles [0..9) or [9..18)
  const int half = lane >> 4;
  const int l16  = lane & 15;

  v8f acc[9];
  const v8f vzero = {};
#pragma unroll
  for (int i = 0; i < 9; ++i) acc[i] = vzero;

  // staging role: thread handles k-pair kk (k=2kk,2kk+1) for pixels p4..p4+3
  const int kk = tid >> 4;                     // 0..15
  const int p4 = (tid & 15) << 2;              // 0,4,...,60
  const float* fbase = F + (size_t)(2 * kk) * HW + hw0 + p4;
  const int pix = nsub * 16 + l16;             // B-fragment pixel column

  // ---- W-fragment staging (TDM or cooperative fallback) ----
#if HAVE_TDM
  const unsigned long long wbase = (unsigned long long)(const void*)wfrag;
  auto tdm_issue = [&](int ks1, int buf) {
    unsigned long long ga = wbase + (unsigned long long)ks1 * WSTEP_BYTES;
    unsigned int lds_off = (unsigned int)(unsigned long long)(void*)&ldsW[buf][0];
    v4u g0 = { 1u,                                   // count=1, user D#
               lds_off,                              // LDS byte address
               (unsigned int)ga,                     // global_addr[31:0]
               (unsigned int)(ga >> 32) | (2u << 30) };  // addr[56:32] | type=2
    v8i g1;
    g1[0] = 0x20000;                 // workgroup_mask=0, data_size=2 (4 B)
    g1[1] = (int)(4608u << 16);      // tensor_dim0[15:0]=4608 (bits 63:48)
    g1[2] = (int)(1u << 16);         // tensor_dim0[31:16]=0 ; tensor_dim1=1
    g1[3] = (int)(4608u << 16);      // tensor_dim1[31:16]=0 ; tile_dim0=4608
    g1[4] = 1;                       // tile_dim1=1 ; tile_dim2=0 (unused)
    g1[5] = 4608;                    // tensor_dim0_stride[31:0]
    g1[6] = 0;                       // stride0[47:32]=0 ; stride1[15:0]=0
    g1[7] = 0;
    v4i z4 = { 0, 0, 0, 0 };         // groups 2/3: higher dims unused
    v8i z8 = { 0, 0, 0, 0, 0, 0, 0, 0 };
    __builtin_amdgcn_tensor_load_to_lds(g0, g1, z4, z4, z8, 0);
  };
#else
  auto coop_copy_w = [&](int ks1, int buf) {   // 1152 x uint4 over 256 threads
    const uint4* src = (const uint4*)(wfrag + (size_t)ks1 * (WSTEP_BYTES / 2));
    uint4* dst = (uint4*)&ldsW[buf][0];
#pragma unroll
    for (int i = 0; i < 5; ++i) {
      int idx = tid + i * 256;
      if (idx < WSTEP_BYTES / 16) dst[idx] = src[idx];
    }
  };
#endif

  auto stage_feat = [&](float4 a0, float4 a1, int buf) {
    unsigned int pk0 = (unsigned int)f32_bf16(a0.x) | ((unsigned int)f32_bf16(a1.x) << 16);
    unsigned int pk1 = (unsigned int)f32_bf16(a0.y) | ((unsigned int)f32_bf16(a1.y) << 16);
    unsigned int pk2 = (unsigned int)f32_bf16(a0.z) | ((unsigned int)f32_bf16(a1.z) << 16);
    unsigned int pk3 = (unsigned int)f32_bf16(a0.w) | ((unsigned int)f32_bf16(a1.w) << 16);
    unsigned short* b = &ldsF[buf][0];
    *(unsigned int*)&b[(p4 + 0) * LDSPITCH + 2 * kk] = pk0;
    *(unsigned int*)&b[(p4 + 1) * LDSPITCH + 2 * kk] = pk1;
    *(unsigned int*)&b[(p4 + 2) * LDSPITCH + 2 * kk] = pk2;
    *(unsigned int*)&b[(p4 + 3) * LDSPITCH + 2 * kk] = pk3;
  };

  // ---------------- prologue: stage K-step 0 ----------------
  {
    float4 a0 = *(const float4*)fbase;
    float4 a1 = *(const float4*)(fbase + HW);
#if HAVE_TDM
    if (wave == 0) tdm_issue(0, 0);
#else
    coop_copy_w(0, 0);
#endif
    stage_feat(a0, a1, 0);
#if HAVE_TDM
    if (wave == 0) __builtin_amdgcn_s_wait_tensorcnt(0);
#endif
    __syncthreads();
  }

  // ---------------- main K loop (one barrier per step) ----------------
  for (int ks = 0; ks < KSTEPS; ++ks) {
    const int cur = ks & 1;
    const int nxt = cur ^ 1;
    const bool more = (ks + 1) < KSTEPS;

    float4 a0, a1;
    if (more) {                                // preload next feature tile
      const float* s1 = fbase + (size_t)(ks + 1) * 32 * HW;
      a0 = *(const float4*)s1;
      a1 = *(const float4*)(s1 + HW);
#if HAVE_TDM
      if (wave == 0) tdm_issue(ks + 1, nxt);   // async TDM, one step ahead
#else
      coop_copy_w(ks + 1, nxt);
#endif
    }

    // ---- B fragment (32x16 K-major): lane holds column pix ----
    const unsigned short* bp = &ldsF[cur][pix * LDSPITCH + half * 8];
    V16B bf;
    bf.q[0] = *(const uint4*)(bp);             // K = half*8 .. half*8+7
    bf.q[1] = *(const uint4*)(bp + 16);        // K = half*8+16 .. +23

    // ---- 9 M-tiles: A fragments from LDS, accumulate with WMMA ----
#pragma unroll
    for (int mt = 0; mt < 9; ++mt) {
      const int tile = mt0 + mt;
      const uint4* ap = (const uint4*)&ldsW[cur][(tile * 32 + lane) * 16];
      V16B af;
      af.q[0] = ap[0];
      af.q[1] = ap[1];
      acc[mt] = __builtin_amdgcn_wmma_f32_16x16x32_bf16(
          false, af.v, false, bf.v, (short)0, acc[mt], false, false);
    }

    if (more) stage_feat(a0, a1, nxt);
#if HAVE_TDM
    if (more && wave == 0) __builtin_amdgcn_s_wait_tensorcnt(0);
#endif
    __syncthreads();
  }

  // ---- epilogue: C/D layout VGPR r -> M = r (+8 for lanes 16..31) ----
  const int col = hw0 + nsub * 16 + l16;
#pragma unroll
  for (int mt = 0; mt < 9; ++mt) {
#pragma unroll
    for (int r = 0; r < 8; ++r) {
      const int row = (mt0 + mt) * 16 + r + half * 8;
      float v = acc[mt][r];
      if (row < 189)                    v += b_cls[row];
      else if (row >= 192 && row < 264) v += b_reg[row - 192];
      else continue;                    // padding rows: never read downstream
      out[(size_t)row * HW + col] = v;
    }
  }
}

// ---------------------------------------------------------------------------
// Kernel 3: position-sensitive RoI pooling. One block per ROI.
// dim d<8  -> reg map row 192 + d*9+i*3+j -> d_out[r*8+d]
// dim d>=8 -> cls map row (d-8)*9+i*3+j   -> d_out[32000 + r*21 + (d-8)]
// ---------------------------------------------------------------------------
__global__ __launch_bounds__(128) void psroi_kernel(
    const float* __restrict__ maps,            // [272][16384] fp32 (ws)
    const float* __restrict__ rois,
    float* __restrict__ d_out) {
  const int r = blockIdx.x;
  __shared__ float sacc[29];
  if (threadIdx.x < 29) sacc[threadIdx.x] = 0.0f;
  __syncthreads();

  const float x1 = rois[r * 4 + 0] * 0.0625f;
  const float y1 = rois[r * 4 + 1] * 0.0625f;
  const float x2 = rois[r * 4 + 2] * 0.0625f;
  const float y2 = rois[r * 4 + 3] * 0.0625f;
  const float bw = fmaxf(x2 - x1, 0.1f) * (1.0f / 3.0f);
  const float bh = fmaxf(y2 - y1, 0.1f) * (1.0f / 3.0f);

  for (int idx = threadIdx.x; idx < 29 * 36; idx += blockDim.x) {
    int s = idx & 3;                           // sample: sy*2+sx
    int t = idx >> 2;
    int j = t % 3; t /= 3;
    int i = t % 3;
    int d = t / 3;                             // 0..28
    int sx = s & 1, sy = s >> 1;

    float xc = x1 + ((float)j + ((float)sx + 0.5f) * 0.5f) * bw;
    float yc = y1 + ((float)i + ((float)sy + 0.5f) * 0.5f) * bh;
    xc = fminf(fmaxf(xc, 0.0f), 127.0f);
    yc = fminf(fmaxf(yc, 0.0f), 127.0f);
    float x0f = fminf(floorf(xc), 126.0f);
    float y0f = fminf(floorf(yc), 126.0f);
    float dx = xc - x0f, dy = yc - y0f;
    int x0 = (int)x0f, y0 = (int)y0f;

    int ch = (d < 8) ? (192 + d * 9 + i * 3 + j) : ((d - 8) * 9 + i * 3 + j);
    const float* p = maps + (size_t)ch * HW + y0 * 128 + x0;
    float v00 = p[0], v01 = p[1], v10 = p[128], v11 = p[129];
    float v = v00 * (1.0f - dy) * (1.0f - dx) + v01 * (1.0f - dy) * dx +
              v10 * dy * (1.0f - dx)          + v11 * dy * dx;
    atomicAdd(&sacc[d], v);                    // ds_add_f32
  }
  __syncthreads();

  if (threadIdx.x < 29) {
    float v = sacc[threadIdx.x] * (1.0f / 36.0f);
    int d = threadIdx.x;
    if (d < 8) d_out[(size_t)r * 8 + d] = v;                    // roi_cls_locs
    else       d_out[32000 + (size_t)r * 21 + (d - 8)] = v;     // roi_scores
  }
}

// ---------------------------------------------------------------------------
extern "C" void kernel_launch(void* const* d_in, const int* in_sizes, int n_in,
                              void* d_out, int out_size, void* d_ws, size_t ws_size,
                              hipStream_t stream) {
  const float* F      = (const float*)d_in[0];   // [1,1024,128,128]
  const float* rois   = (const float*)d_in[1];   // [4000,4]
  // d_in[2] = roi_indices (all zeros, single batch) -- unused
  const float* W_cls  = (const float*)d_in[3];   // [189,1024]
  const float* b_cls  = (const float*)d_in[4];   // [189]
  const float* W_reg  = (const float*)d_in[5];   // [72,1024]
  const float* b_reg  = (const float*)d_in[6];   // [72]
  float* out = (float*)d_out;                    // 4000*8 locs ++ 4000*21 scores

  // workspace layout
  float* maps = (float*)d_ws;                                        // 272*16384 f32
  unsigned short* wfrag =
      (unsigned short*)((char*)d_ws + (size_t)272 * HW * sizeof(float));

  {
    int nthreads = TILES * KSTEPS * 32;                              // 18432
    prep_wfrag<<<(nthreads + 255) / 256, 256, 0, stream>>>(W_cls, W_reg, wfrag);
  }
  fused_gemm<<<HW / 64, 256, 0, stream>>>(F, wfrag, b_cls, b_reg, maps);
  psroi_kernel<<<4000, 128, 0, stream>>>(maps, rois, out);
}
